// GFlowGNN_9388798509588
// MI455X (gfx1250) — matrix operand order repaired and checked by reading, not compiled, not run on hardware
//
#include <hip/hip_runtime.h>
#include <hip/hip_bf16.h>
#include <stdint.h>

typedef __attribute__((ext_vector_type(16))) _Float16 v16h;
typedef __attribute__((ext_vector_type(8)))  float    v8f;
typedef __attribute__((ext_vector_type(8)))  uint32_t v8u;

// ---------------------------------------------------------------------------
// Degree / normalization kernels
// ---------------------------------------------------------------------------
__global__ void deg_init_kernel(float* __restrict__ deg, int n) {
    int i = blockIdx.x * blockDim.x + threadIdx.x;
    if (i < n) deg[i] = 1.0f;  // self loop
}

__global__ void deg_edges_kernel(const int* __restrict__ dst,
                                 float* __restrict__ deg, int e) {
    int i = blockIdx.x * blockDim.x + threadIdx.x;
    if (i < e) atomicAdd(&deg[dst[i]], 1.0f);
}

__global__ void make_dis_kernel(float* __restrict__ degdis, int n) {
    int i = blockIdx.x * blockDim.x + threadIdx.x;
    if (i < n) degdis[i] = rsqrtf(degdis[i]);  // deg >= 1 guaranteed
}

// ---------------------------------------------------------------------------
// WMMA GEMM: Out[N,64] = (RELU? max(A,0) : A)[N,64] @ W[64,64]
// One wave -> 16 rows x 64 cols. 8 waves/block -> 128 rows/block.
// W is pre-packed into LDS in the exact B-fragment VGPR layout.
// RELU is a compile-time template parameter so the A-fragment build has no
// select chains (layer 1: plain cvt; layer 2: one v_max per element).
// ---------------------------------------------------------------------------
template <int RELU>
__global__ void __launch_bounds__(256)
gemm64_wmma_kernel(const float* __restrict__ A,
                   const float* __restrict__ W,
                   float* __restrict__ Out,
                   int nrows) {
    // Bs[kc][nt][v][lane] : packed f16 pair {K=base+2v, K=base+2v+1} of column n
    __shared__ uint32_t Bs[2][4][8][32];  // 8 KB

    const int t = threadIdx.x;
    // Cooperative pack of W (f32 -> f16 pairs) into B-fragment order.
    for (int idx = t; idx < 2048; idx += 256) {
        int lane = idx & 31;
        int vi   = (idx >> 5) & 7;
        int nt   = (idx >> 8) & 3;
        int kc   = (idx >> 10) & 1;
        int n = nt * 16 + (lane & 15);
        int k = kc * 32 + ((vi < 4) ? ((lane >> 4) * 8 + 2 * vi) : (16 + (lane >> 4) * 8 + 2 * (vi - 4)));   // lanes 0-15: K 0-7,16-23, lanes 16-31: K 8-15,24-31
        _Float16 lo = (_Float16)W[(size_t)k * 64 + n];
        _Float16 hi = (_Float16)W[(size_t)(k + 1) * 64 + n];
        uint32_t p = (uint32_t)__builtin_bit_cast(uint16_t, lo) |
                     ((uint32_t)__builtin_bit_cast(uint16_t, hi) << 16);
        Bs[kc][nt][vi][lane] = p;
    }
    __syncthreads();

    const int wave = t >> 5;
    const int lane = t & 31;
    const int row0 = (blockIdx.x * 8 + wave) * 16;
    if (row0 >= nrows) return;  // uniform per wave -> EXEC stays all-ones for WMMA

    // ---- A fragments (ISA 16-bit A layout: lane=row, K split across lane halves)
    const int m  = lane & 15;
    const int kb = (lane >> 4) * 8;  // 0 or 8
    const float* arow = A + (size_t)(row0 + m) * 64;

    v16h afrag[2];
    #pragma unroll
    for (int kc = 0; kc < 2; ++kc) {
        #pragma unroll
        for (int j = 0; j < 8; ++j) {
            int k = kc * 32 + ((j < 4) ? (kb + 2 * j) : (16 + kb + 2 * (j - 4)));
            float x0 = arow[k];
            float x1 = arow[k + 1];
            if (RELU) { x0 = fmaxf(x0, 0.0f); x1 = fmaxf(x1, 0.0f); }
            afrag[kc][2 * j]     = (_Float16)x0;
            afrag[kc][2 * j + 1] = (_Float16)x1;
        }
    }

    // ---- 4 N-tiles, accumulate over K=64 in two K=32 WMMAs each
    #pragma unroll
    for (int nt = 0; nt < 4; ++nt) {
        v8f acc = {};
        #pragma unroll
        for (int kc = 0; kc < 2; ++kc) {
            v8u bv;
            #pragma unroll
            for (int vi = 0; vi < 8; ++vi) bv[vi] = Bs[kc][nt][vi][lane];
            v16h bfrag = __builtin_bit_cast(v16h, bv);
            acc = __builtin_amdgcn_wmma_f32_16x16x32_f16(
                false, afrag[kc], false, bfrag, (short)0, acc, false, false);
        }
        // ---- D layout: VGPR r -> lanes 0-15: (M=r, N=lane) ; lanes 16-31: (M=r+8, N=lane-16)
        const int ms = (lane >> 4) * 8;
        const int n  = nt * 16 + (lane & 15);
        #pragma unroll
        for (int r = 0; r < 8; ++r) {
            Out[(size_t)(row0 + ms + r) * 64 + n] = acc[r];
        }
    }
}

// ---------------------------------------------------------------------------
// Aggregation: out[i] = xw[i]*dis[i]^2 + b   (self loop + bias, full overwrite)
// One thread per (node, float4 chunk): N*16 threads.
// ---------------------------------------------------------------------------
__global__ void agg_init_kernel(const float* __restrict__ xw,
                                const float* __restrict__ dis,
                                const float* __restrict__ bias,
                                float* __restrict__ out, int n) {
    unsigned tid = blockIdx.x * blockDim.x + threadIdx.x;
    if (tid >= (unsigned)n * 16u) return;
    int i = tid >> 4, q = tid & 15;
    float s = dis[i];
    s = s * s;
    float4 v = ((const float4*)xw)[(size_t)i * 16 + q];
    float4 bb = ((const float4*)bias)[q];
    float4 o;
    o.x = v.x * s + bb.x;
    o.y = v.y * s + bb.y;
    o.z = v.z * s + bb.z;
    o.w = v.w * s + bb.w;
    ((float4*)out)[(size_t)i * 16 + q] = o;
}

// ---------------------------------------------------------------------------
// Edge messages: out[dst] += xw[src] * dis[src]*dis[dst]
// One thread per (edge, float4 chunk): E*16 threads; f32 atomics land in L2
// (out + xw arrays are 25.6 MB each -> resident in the 192 MB L2).
// ---------------------------------------------------------------------------
__global__ void agg_edges_kernel(const int* __restrict__ src,
                                 const int* __restrict__ dst,
                                 const float* __restrict__ xw,
                                 const float* __restrict__ dis,
                                 float* __restrict__ out, int e) {
    unsigned tid = blockIdx.x * blockDim.x + threadIdx.x;
    if (tid >= (unsigned)e * 16u) return;
    int ei = tid >> 4, q = tid & 15;
    int s = src[ei];
    int d = dst[ei];
    float norm = dis[s] * dis[d];
    float4 v = ((const float4*)xw)[(size_t)s * 16 + q];
    float* o = out + (size_t)d * 64 + q * 4;
    atomicAdd(o + 0, v.x * norm);
    atomicAdd(o + 1, v.y * norm);
    atomicAdd(o + 2, v.z * norm);
    atomicAdd(o + 3, v.w * norm);
}

// ---------------------------------------------------------------------------
// Launcher
// ---------------------------------------------------------------------------
extern "C" void kernel_launch(void* const* d_in, const int* in_sizes, int n_in,
                              void* d_out, int out_size, void* d_ws, size_t ws_size,
                              hipStream_t stream) {
    const float* x  = (const float*)d_in[0];
    const int*   ei = (const int*)d_in[1];
    const float* W1 = (const float*)d_in[2];
    const float* b1 = (const float*)d_in[3];
    const float* W2 = (const float*)d_in[4];
    const float* b2 = (const float*)d_in[5];

    const int N = in_sizes[0] / 64;
    const int E = in_sizes[1] / 2;
    const int* src = ei;
    const int* dst = ei + E;

    // Workspace layout (256B aligned): dis[N] | xw[N*64] | h[N*64]  (~52 MB)
    char* ws = (char*)d_ws;
    size_t off = 0;
    float* dis = (float*)(ws + off); off += (((size_t)N * 4) + 255) & ~(size_t)255;
    float* xw  = (float*)(ws + off); off += (((size_t)N * 64 * 4) + 255) & ~(size_t)255;
    float* h   = (float*)(ws + off);
    float* out = (float*)d_out;

    const int TB = 256;
    dim3 blk(TB);

    // ---- degree + symmetric normalization (once, reused by both layers)
    deg_init_kernel<<<dim3((N + TB - 1) / TB), blk, 0, stream>>>(dis, N);
    deg_edges_kernel<<<dim3((E + TB - 1) / TB), blk, 0, stream>>>(dst, dis, E);
    make_dis_kernel<<<dim3((N + TB - 1) / TB), blk, 0, stream>>>(dis, N);

    const int ntiles     = (N + 15) / 16;
    const int gemm_blks  = (ntiles + 7) / 8;
    const unsigned nwork = (unsigned)N * 16u;
    const unsigned ework = (unsigned)E * 16u;

    // ---- layer 1: h = agg(x @ W1) + b1   (ReLU deferred to layer-2 A-load)
    gemm64_wmma_kernel<0><<<dim3(gemm_blks), blk, 0, stream>>>(x, W1, xw, N);
    agg_init_kernel<<<dim3((nwork + TB - 1) / TB), blk, 0, stream>>>(xw, dis, b1, h, N);
    agg_edges_kernel<<<dim3((ework + TB - 1) / TB), blk, 0, stream>>>(src, dst, xw, dis, h, E);

    // ---- layer 2: out = agg(relu(h) @ W2) + b2
    gemm64_wmma_kernel<1><<<dim3(gemm_blks), blk, 0, stream>>>(h, W2, xw, N);
    agg_init_kernel<<<dim3((nwork + TB - 1) / TB), blk, 0, stream>>>(xw, dis, b2, out, N);
    agg_edges_kernel<<<dim3((ework + TB - 1) / TB), blk, 0, stream>>>(src, dst, xw, dis, out, E);
}
